// SDEModel2Dto3D_01_73031623901268
// MI455X (gfx1250) — compile-verified, matrix operand order checked by reading, not run on hardware
//
#include <hip/hip_runtime.h>
#include <hip/hip_bf16.h>
#include <math.h>

#define N_NODES 10000
#define N_EDGES 160000
#define HID 128
#define HEADS 8
#define HEAD_D 16
#define COFF 64
#define QKVS 512   // Wq|Wk|Wv|Ws packed

typedef __bf16 bf16_t;
typedef __attribute__((ext_vector_type(16))) __bf16 v16bf;
typedef __attribute__((ext_vector_type(8)))  float  v8f;

// fast silu: v * rcp(1+exp(-v)) -> v_exp_f32 + v_rcp_f32, no IEEE divide
__device__ __forceinline__ float silu_f(float v) {
  return v * __builtin_amdgcn_rcpf(1.f + __expf(-v));
}

// Convert two contiguous 8-float spans (float4 x4) into a CDNA5 16-bit A fragment.
// Span mapping per ISA: frag[0..7] <- A[kb+8h .. +7], frag[8..15] <- A[kb+16+8h .. +7].
__device__ __forceinline__ v16bf frag_from_spans(float4 f0, float4 f1, float4 f2, float4 f3) {
  v16bf a;
  a[0]=(__bf16)f0.x; a[1]=(__bf16)f0.y; a[2]=(__bf16)f0.z; a[3]=(__bf16)f0.w;
  a[4]=(__bf16)f1.x; a[5]=(__bf16)f1.y; a[6]=(__bf16)f1.z; a[7]=(__bf16)f1.w;
  a[8]=(__bf16)f2.x; a[9]=(__bf16)f2.y; a[10]=(__bf16)f2.z; a[11]=(__bf16)f2.w;
  a[12]=(__bf16)f3.x; a[13]=(__bf16)f3.y; a[14]=(__bf16)f3.z; a[15]=(__bf16)f3.w;
  return a;
}

// ---------------------------------------------------------------------------
// GEMM: C[rows,NCOLS] = act(A[rows,K](f32->bf16) @ W + bias)
// W pre-swizzled into per-lane fragments: Wz[((kt*ntiles+nt)*32+lane)*16 + j]
// One wave computes 16 rows x 64 cols (4 N-tiles): one A fragment reused by 4
// WMMAs. K/NCOLS compile-time -> full unroll, div/mod becomes shift/mask.
// ---------------------------------------------------------------------------
template<int K, int NCOLS>
__global__ void k_gemm_t(const float* __restrict__ A, const bf16_t* __restrict__ Wz,
                         const float* __restrict__ bias, float* __restrict__ C,
                         int rows, int act)
{
  constexpr int NTILES = NCOLS >> 4;
  constexpr int GROUPS = NTILES >> 2;                   // power of two
  constexpr int KTILES = K >> 5;
  const int lane = threadIdx.x;
  const int half = lane >> 4;
  const int lm   = lane & 15;
  const int wid  = blockIdx.x * blockDim.y + threadIdx.y;
  const int rt   = wid / GROUPS;                        // row tile (shift)
  const int g    = wid % GROUPS;                        // group of 4 n-tiles (mask)
  if (rt >= (rows >> 4)) return;                        // uniform per wave
  const int nt0  = g << 2;
  const int row0 = rt << 4;
  int arow = row0 + lm; if (arow >= rows) arow = rows - 1;
  const float4* __restrict__ Ar4 = (const float4*)(A + (unsigned)(arow * K));
  v8f cc[4];
#pragma unroll
  for (int j = 0; j < 4; ++j) cc[j] = (v8f){0.f,0.f,0.f,0.f,0.f,0.f,0.f,0.f};
#pragma unroll
  for (int kt = 0; kt < KTILES; ++kt) {
    const int b0i = (kt << 3) + (half << 1);            // float4 index of kb + 8*half
    const float4 f0 = Ar4[b0i],     f1 = Ar4[b0i + 1];
    const float4 f2 = Ar4[b0i + 4], f3 = Ar4[b0i + 5];
    const v16bf* __restrict__ Wf =
        (const v16bf*)(Wz + ((unsigned)((kt * NTILES + nt0) * 32 + lane) << 4));
    const v16bf b0 = Wf[0], b1 = Wf[32], b2 = Wf[64], b3 = Wf[96];
    const v16bf a = frag_from_spans(f0, f1, f2, f3);
    cc[0] = __builtin_amdgcn_wmma_f32_16x16x32_bf16(false, a, false, b0, (short)0, cc[0], false, false);
    cc[1] = __builtin_amdgcn_wmma_f32_16x16x32_bf16(false, a, false, b1, (short)0, cc[1], false, false);
    cc[2] = __builtin_amdgcn_wmma_f32_16x16x32_bf16(false, a, false, b2, (short)0, cc[2], false, false);
    cc[3] = __builtin_amdgcn_wmma_f32_16x16x32_bf16(false, a, false, b3, (short)0, cc[3], false, false);
  }
#pragma unroll
  for (int j = 0; j < 4; ++j) {
    const int n = ((nt0 + j) << 4) + lm;
    const float bv = bias ? bias[n] : 0.f;
#pragma unroll
    for (int r = 0; r < 8; ++r) {
      const int row = row0 + r + (half << 3);
      if (row < rows) {
        float v = cc[j][r] + bv;
        if (act) v = silu_f(v);
        C[(unsigned)(row * NCOLS) + n] = v;
      }
    }
  }
}

// ---------------------------------------------------------------------------
// Basis MLP first layer, fused gather (never materializes ef[E,256]):
// ef[e,0:128] = x[src]+x[dst] ; ef[e,128:256] = edge_attr[e]
// C[e,0:64] = silu(ef @ W1 + b1), K=256. One wave does 16 edges x all 64 cols,
// so the 2-way gather happens once per edge-tile.
// ---------------------------------------------------------------------------
__global__ void k_basis_gemm(const float* __restrict__ x, const float* __restrict__ ea,
                             const int* __restrict__ src, const int* __restrict__ dst,
                             const bf16_t* __restrict__ Wz, const float* __restrict__ bias,
                             float* __restrict__ C)
{
  const int lane = threadIdx.x, half = lane >> 4, lm = lane & 15;
  const int row0 = ((blockIdx.x << 2) + threadIdx.y) << 4;   // 4 waves -> 4 edge tiles
  int e = row0 + lm; if (e >= N_EDGES) e = N_EDGES - 1;
  const int sj = src[e], dj = dst[e];
  const float4* __restrict__ xs4 = (const float4*)(x  + (unsigned)(sj * HID));
  const float4* __restrict__ xd4 = (const float4*)(x  + (unsigned)(dj * HID));
  const float4* __restrict__ er4 = (const float4*)(ea + (unsigned)(e  * HID));
  v8f cc[4];
#pragma unroll
  for (int j = 0; j < 4; ++j) cc[j] = (v8f){0.f,0.f,0.f,0.f,0.f,0.f,0.f,0.f};
#pragma unroll
  for (int kt = 0; kt < 8; ++kt) {
    float4 f0, f1, f2, f3;
    if (kt < 4) {                                  // uniform branch: x[src]+x[dst]
      const int b0i = (kt << 3) + (half << 1);
      float4 p0 = xs4[b0i], p1 = xs4[b0i+1], p2 = xs4[b0i+4], p3 = xs4[b0i+5];
      float4 q0 = xd4[b0i], q1 = xd4[b0i+1], q2 = xd4[b0i+4], q3 = xd4[b0i+5];
      f0.x=p0.x+q0.x; f0.y=p0.y+q0.y; f0.z=p0.z+q0.z; f0.w=p0.w+q0.w;
      f1.x=p1.x+q1.x; f1.y=p1.y+q1.y; f1.z=p1.z+q1.z; f1.w=p1.w+q1.w;
      f2.x=p2.x+q2.x; f2.y=p2.y+q2.y; f2.z=p2.z+q2.z; f2.w=p2.w+q2.w;
      f3.x=p3.x+q3.x; f3.y=p3.y+q3.y; f3.z=p3.z+q3.z; f3.w=p3.w+q3.w;
    } else {                                       // edge_attr half
      const int b0i = ((kt - 4) << 3) + (half << 1);
      f0 = er4[b0i]; f1 = er4[b0i+1]; f2 = er4[b0i+4]; f3 = er4[b0i+5];
    }
    const v16bf* __restrict__ Wf = (const v16bf*)(Wz + ((unsigned)(kt * 128 + lane) << 4));
    const v16bf b0 = Wf[0], b1 = Wf[32], b2 = Wf[64], b3 = Wf[96];
    const v16bf a = frag_from_spans(f0, f1, f2, f3);
    cc[0] = __builtin_amdgcn_wmma_f32_16x16x32_bf16(false, a, false, b0, (short)0, cc[0], false, false);
    cc[1] = __builtin_amdgcn_wmma_f32_16x16x32_bf16(false, a, false, b1, (short)0, cc[1], false, false);
    cc[2] = __builtin_amdgcn_wmma_f32_16x16x32_bf16(false, a, false, b2, (short)0, cc[2], false, false);
    cc[3] = __builtin_amdgcn_wmma_f32_16x16x32_bf16(false, a, false, b3, (short)0, cc[3], false, false);
  }
#pragma unroll
  for (int j = 0; j < 4; ++j) {
    const int n = (j << 4) + lm;
    const float bv = bias[n];
#pragma unroll
    for (int r = 0; r < 8; ++r) {
      const int row = row0 + r + (half << 3);
      if (row < N_EDGES) C[(unsigned)(row * COFF) + n] = silu_f(cc[j][r] + bv);
    }
  }
}

// ---------------------------------------------------------------------------
// Attention: one wave per destination node, online softmax over CSR edges,
// single pass, no atomics. lane -> 4 channels, head = lane/4.
// epair[ii] = {edge_id, src_node} packed at CSR build time (one b64 per edge).
// qkvs row: [q|k|v|skip] stride 512.
// ---------------------------------------------------------------------------
__global__ void k_attn(const float* __restrict__ qkvs, const float* __restrict__ eproj,
                       const int* __restrict__ rowptr, const int2* __restrict__ epair,
                       float* __restrict__ h)
{
  const int node = blockIdx.x * blockDim.y + threadIdx.y;
  if (node >= N_NODES) return;
  const int lane = threadIdx.x;
  const float4 qv = ((const float4*)(qkvs + (unsigned)(node * QKVS)))[lane];
  float m = -__builtin_inff(), ssum = 0.f;
  float4 acc = make_float4(0.f, 0.f, 0.f, 0.f);
  const int beg = rowptr[node], end = rowptr[node + 1];
  for (int ii = beg; ii < end; ++ii) {
    const int2 pr = epair[ii];
    const int e = pr.x, j = pr.y;
    if (ii + 1 < end) {                       // hide gather latency
      const int2 p2 = epair[ii + 1];
      __builtin_prefetch(eproj + (unsigned)(p2.x * HID) + lane * 4, 0, 0);
      __builtin_prefetch(qkvs + (unsigned)(p2.y * QKVS) + HID + lane * 4, 0, 0);
    }
    const float4 ep = ((const float4*)(eproj + (unsigned)(e * HID)))[lane];
    float4 kj = ((const float4*)(qkvs + (unsigned)(j * QKVS) + HID))[lane];
    float4 vj = ((const float4*)(qkvs + (unsigned)(j * QKVS) + 2 * HID))[lane];
    kj.x += ep.x; kj.y += ep.y; kj.z += ep.z; kj.w += ep.w;
    vj.x += ep.x; vj.y += ep.y; vj.z += ep.z; vj.w += ep.w;
    float t = qv.x * kj.x + qv.y * kj.y + qv.z * kj.z + qv.w * kj.w;
    t += __shfl_xor(t, 1, 32);
    t += __shfl_xor(t, 2, 32);                 // 16-dim head = 4 lanes
    const float logit = t * 0.25f;             // 1/sqrt(HEAD_D)
    const float nm   = fmaxf(m, logit);
    const float corr = __expf(m - nm);         // first iter: exp(-inf)=0
    const float w    = __expf(logit - nm);
    ssum  = ssum * corr + w;
    acc.x = acc.x * corr + w * vj.x;
    acc.y = acc.y * corr + w * vj.y;
    acc.z = acc.z * corr + w * vj.z;
    acc.w = acc.w * corr + w * vj.w;
    m = nm;
  }
  const float inv = __builtin_amdgcn_rcpf(ssum + 1e-16f);
  const float4 sk = ((const float4*)(qkvs + (unsigned)(node * QKVS) + 3 * HID))[lane];
  float4 o;
  o.x = acc.x * inv + sk.x; o.y = acc.y * inv + sk.y;
  o.z = acc.z * inv + sk.z; o.w = acc.w * inv + sk.w;
  ((float4*)(h + (unsigned)(node * HID)))[lane] = o;
}

// x = x + layernorm(h)*g + be ; optional trailing silu. One wave per node.
__global__ void k_addnorm(float* __restrict__ x, const float* __restrict__ h,
                          const float* __restrict__ g, const float* __restrict__ be,
                          int post_silu)
{
  const int node = blockIdx.x * blockDim.y + threadIdx.y;
  if (node >= N_NODES) return;
  const int lane = threadIdx.x;
  const float4 hv = ((const float4*)(h + (unsigned)(node * HID)))[lane];
  float s = hv.x + hv.y + hv.z + hv.w;
#pragma unroll
  for (int off = 1; off < 32; off <<= 1) s += __shfl_xor(s, off, 32);
  const float mean = s * (1.f / HID);
  float4 d;
  d.x = hv.x - mean; d.y = hv.y - mean; d.z = hv.z - mean; d.w = hv.w - mean;
  float vs = d.x * d.x + d.y * d.y + d.z * d.z + d.w * d.w;
#pragma unroll
  for (int off = 1; off < 32; off <<= 1) vs += __shfl_xor(vs, off, 32);
  const float inv = rsqrtf(vs * (1.f / HID) + 1e-5f);
  const float4 xv = ((const float4*)(x + (unsigned)(node * HID)))[lane];
  const float4 gv = ((const float4*)g)[lane];
  const float4 bv = ((const float4*)be)[lane];
  float4 o;
  o.x = xv.x + d.x * inv * gv.x + bv.x;
  o.y = xv.y + d.y * inv * gv.y + bv.y;
  o.z = xv.z + d.z * inv * gv.z + bv.z;
  o.w = xv.w + d.w * inv * gv.w + bv.w;
  if (post_silu) { o.x = silu_f(o.x); o.y = silu_f(o.y); o.z = silu_f(o.z); o.w = silu_f(o.w); }
  ((float4*)(x + (unsigned)(node * HID)))[lane] = o;
}

// ----------------------------- CSR build -----------------------------------
__global__ void k_count(const int* __restrict__ dst, int* __restrict__ counts)
{
  const int e = blockIdx.x * blockDim.x + threadIdx.x;
  if (e < N_EDGES) atomicAdd(&counts[dst[e]], 1);
}

__global__ void k_scan(const int* __restrict__ counts, int* __restrict__ rowptr,
                       float* __restrict__ cntf)
{
  __shared__ int part[1024];
  const int t = threadIdx.x;
  const int CHUNK = (N_NODES + 1023) / 1024;
  const int base = t * CHUNK;
  int local[(N_NODES + 1023) / 1024];
  int s = 0;
#pragma unroll
  for (int i = 0; i < CHUNK; ++i) {
    const int idx = base + i;
    const int c = (idx < N_NODES) ? counts[idx] : 0;
    local[i] = s; s += c;
  }
  part[t] = s; __syncthreads();
  for (int off = 1; off < 1024; off <<= 1) {
    const int v = (t >= off) ? part[t - off] : 0;
    __syncthreads();
    part[t] += v;
    __syncthreads();
  }
  const int pre = (t == 0) ? 0 : part[t - 1];
#pragma unroll
  for (int i = 0; i < CHUNK; ++i) {
    const int idx = base + i;
    if (idx < N_NODES) {
      rowptr[idx] = pre + local[i];
      cntf[idx]   = fmaxf((float)counts[idx], 1.f);
    }
  }
  if (t == 1023) rowptr[N_NODES] = part[1023];
}

// Scatter edges into CSR order as packed {edge_id, src_node} records.
__global__ void k_scatter(const int* __restrict__ src, const int* __restrict__ dst,
                          const int* __restrict__ rowptr, int* __restrict__ fill,
                          int2* __restrict__ epair)
{
  const int e = blockIdx.x * blockDim.x + threadIdx.x;
  if (e < N_EDGES) {
    const int d = dst[e];
    const int p = atomicAdd(&fill[d], 1);
    epair[rowptr[d] + p] = make_int2(e, src[e]);
  }
}

// ----------------------------- coord basis ---------------------------------
__global__ void k_coord(const float* __restrict__ pos, const int* __restrict__ src,
                        const int* __restrict__ dst, float* __restrict__ cd,
                        float* __restrict__ cc, float* __restrict__ cv)
{
  const int e = blockIdx.x * blockDim.x + threadIdx.x;
  if (e >= N_EDGES) return;
  const int s = src[e], d = dst[e];
  const float sx = pos[s*3], sy = pos[s*3+1], sz = pos[s*3+2];
  const float tx = pos[d*3], ty = pos[d*3+1], tz = pos[d*3+2];
  float ddx = sx - tx, ddy = sy - ty, ddz = sz - tz;
  const float radial = ddx*ddx + ddy*ddy + ddz*ddz;
  float cx = sy*tz - sz*ty, cy = sz*tx - sx*tz, cz = sx*ty - sy*tx;
  const float rn = __builtin_amdgcn_rcpf(sqrtf(radial) + 1e-6f);
  ddx *= rn; ddy *= rn; ddz *= rn;
  const float cn = __builtin_amdgcn_rcpf(sqrtf(cx*cx + cy*cy + cz*cz) + 1e-6f);
  cx *= cn; cy *= cn; cz *= cn;
  const float vx = ddy*cz - ddz*cy, vy = ddz*cx - ddx*cz, vz = ddx*cy - ddy*cx;
  cd[e*3] = ddx; cd[e*3+1] = ddy; cd[e*3+2] = ddz;
  cc[e*3] = cx;  cc[e*3+1] = cy;  cc[e*3+2] = cz;
  cv[e*3] = vx;  cv[e*3+1] = vy;  cv[e*3+2] = vz;
}

// coff = ehid @ W2 + b2 (64x3, staged in LDS) ; mix with basis ; scatter-add.
__global__ void k_basis_out(const float* __restrict__ ehid, const float* __restrict__ W2,
                            const float* __restrict__ b2,
                            const float* __restrict__ cd, const float* __restrict__ cc,
                            const float* __restrict__ cv, const int* __restrict__ dst,
                            float* __restrict__ grad)
{
  __shared__ float sW2[COFF * 3 + 3];
  for (int i = threadIdx.x; i < COFF * 3; i += blockDim.x) sW2[i] = W2[i];
  if (threadIdx.x < 3) sW2[COFF * 3 + threadIdx.x] = b2[threadIdx.x];
  __syncthreads();
  const int e = blockIdx.x * blockDim.x + threadIdx.x;
  if (e >= N_EDGES) return;
  float c0 = sW2[COFF*3], c1 = sW2[COFF*3+1], c2 = sW2[COFF*3+2];
  const float* hr = ehid + (unsigned)(e * COFF);
#pragma unroll 8
  for (int j = 0; j < COFF; ++j) {
    const float hv = hr[j];
    c0 += hv * sW2[j*3];
    c1 += hv * sW2[j*3 + 1];
    c2 += hv * sW2[j*3 + 2];
  }
  const float gx = c0*cd[e*3]   + c1*cc[e*3]   + c2*cv[e*3];
  const float gy = c0*cd[e*3+1] + c1*cc[e*3+1] + c2*cv[e*3+1];
  const float gz = c0*cd[e*3+2] + c1*cc[e*3+2] + c2*cv[e*3+2];
  const int d = dst[e];
  atomicAdd(&grad[d*3],     gx);
  atomicAdd(&grad[d*3 + 1], gy);
  atomicAdd(&grad[d*3 + 2], gz);
}

__global__ void k_gradout(const float* __restrict__ grad, const float* __restrict__ cntf,
                          float* __restrict__ out)
{
  const int i = blockIdx.x * blockDim.x + threadIdx.x;
  if (i < N_NODES * 3) out[i] = grad[i] / cntf[i / 3];
}

// ------------------------- weight prep -------------------------------------
// Concatenate Wq|Wk|Wv|Ws -> f32 [128,512] + bias [512]
__global__ void k_pack_qkvs_f32(const float* __restrict__ Wq, const float* __restrict__ Wk,
                                const float* __restrict__ Wv, const float* __restrict__ Ws,
                                const float* __restrict__ bq, const float* __restrict__ bk,
                                const float* __restrict__ bv, const float* __restrict__ bs,
                                float* __restrict__ Wout, float* __restrict__ bout)
{
  const int idx = blockIdx.x * blockDim.x + threadIdx.x;
  if (idx < HID * QKVS) {
    const int col = idx & (QKVS - 1);
    const int kk  = idx >> 9;
    const int grp = col >> 7;
    const int c   = col & (HID - 1);
    const float* Wsrc = (grp == 0) ? Wq : (grp == 1) ? Wk : (grp == 2) ? Wv : Ws;
    Wout[idx] = Wsrc[kk * HID + c];
  }
  if (idx < QKVS) {
    const int grp = idx >> 7;
    const int c   = idx & (HID - 1);
    const float* bsrc = (grp == 0) ? bq : (grp == 1) ? bk : (grp == 2) ? bv : bs;
    bout[idx] = bsrc[c];
  }
}

// Swizzle W[K,ncols] f32 into per-lane bf16 WMMA B fragments:
// out[((kt*ntiles+nt)*32 + lane)*16 + {2i,2i+1}] = W[kt*32+koff(i,half)][nt*16+lane%16]
__global__ void k_swizzle_w(const float* __restrict__ W, bf16_t* __restrict__ out,
                            int K, int ncols)
{
  const int idx = blockIdx.x * blockDim.x + threadIdx.x;
  const int ktiles = K >> 5, ntiles = ncols >> 4;
  if (idx >= ktiles * ntiles * 32) return;
  const int lane = idx & 31;
  const int tnk  = idx >> 5;
  const int nt   = tnk % ntiles;
  const int kt   = tnk / ntiles;
  const int half = lane >> 4;
  const int n    = (nt << 4) + (lane & 15);
  v16bf frag;
#pragma unroll
  for (int i = 0; i < 8; ++i) {
    const int ko = ((i & 4) << 2) | ((i & 3) << 1) | (half << 3);
    const int kk = (kt << 5) + ko;
    frag[2*i]   = (__bf16)W[kk * ncols + n];
    frag[2*i+1] = (__bf16)W[(kk + 1) * ncols + n];
  }
  *(v16bf*)(out + ((unsigned)idx << 4)) = frag;
}

// ===========================================================================
extern "C" void kernel_launch(void* const* d_in, const int* in_sizes, int n_in,
                              void* d_out, int out_size, void* d_ws, size_t ws_size,
                              hipStream_t stream)
{
  (void)in_sizes; (void)n_in; (void)out_size; (void)ws_size;
  const float* node_attr = (const float*)d_in[0];
  const float* edge_attr = (const float*)d_in[1];
  const float* pos       = (const float*)d_in[2];
  // setup_inputs() insertion order: gat[l] = {Wq,bq,Wk,bk,Wv,bv,We,Ws,bs,W1,b1,W2,b2,g1,be1,g2,be2}
#define GAT(l, j) ((const float*)d_in[3 + (l)*17 + (j)])
#define BAS(m, j) ((const float*)d_in[3 + 4*17 + (m)*4 + (j)])   // {W1,b1,W2,b2}
  const int* eidx = (const int*)d_in[3 + 4*17 + 2*4];
  const int* src  = eidx;
  const int* dstv = eidx + N_EDGES;

  char* base = (char*)d_ws;
  size_t off = 0;
  auto carve = [&](size_t bytes) -> char* {
    char* p = base + off;
    off = (off + bytes + 255) & ~(size_t)255;
    return p;
  };
  int*   counts = (int*)carve((size_t)N_NODES * 4);
  int*   fill   = (int*)carve((size_t)N_NODES * 4);
  int*   rowptr = (int*)carve((size_t)(N_NODES + 1) * 4);
  int2*  epair  = (int2*)carve((size_t)N_EDGES * 8);
  float* cntf   = (float*)carve((size_t)N_NODES * 4);
  float* cdiff  = (float*)carve((size_t)N_EDGES * 3 * 4);
  float* ccross = (float*)carve((size_t)N_EDGES * 3 * 4);
  float* cvert  = (float*)carve((size_t)N_EDGES * 3 * 4);
  float* grad   = (float*)carve((size_t)N_NODES * 3 * 4);
  float* xbuf   = (float*)carve((size_t)N_NODES * HID * 4);
  float* qkvs   = (float*)carve((size_t)N_NODES * QKVS * 4);
  float* hbuf   = (float*)carve((size_t)N_NODES * HID * 4);
  float* tbuf   = (float*)carve((size_t)N_NODES * HID * 4);
  float* eproj  = (float*)carve((size_t)N_EDGES * HID * 4);   // reused as basis hidden [E,64]
  float* wtmp   = (float*)carve((size_t)HID * QKVS * 4);      // qkvs concat staging
  bf16_t* wqkvs[4]; float* bqkvs[4]; bf16_t* wwe[4]; bf16_t* ww1[4]; bf16_t* ww2[4];
  for (int l = 0; l < 4; ++l) {
    wqkvs[l] = (bf16_t*)carve((size_t)HID * QKVS * 2);
    bqkvs[l] = (float*)carve((size_t)QKVS * 4);
    wwe[l]   = (bf16_t*)carve((size_t)HID * HID * 2);
    ww1[l]   = (bf16_t*)carve((size_t)HID * HID * 2);
    ww2[l]   = (bf16_t*)carve((size_t)HID * HID * 2);
  }
  bf16_t* wb1[2];
  for (int m = 0; m < 2; ++m) wb1[m] = (bf16_t*)carve((size_t)(2 * HID) * COFF * 2);

  hipMemsetAsync(counts, 0, (size_t)N_NODES * 4, stream);
  hipMemsetAsync(fill,   0, (size_t)N_NODES * 4, stream);
  hipMemsetAsync(grad,   0, (size_t)N_NODES * 3 * 4, stream);

  // CSR + geometry (dst fixed across layers -> build once)
  k_count  <<<(N_EDGES + 255) / 256, 256, 0, stream>>>(dstv, counts);
  k_scan   <<<1, 1024, 0, stream>>>(counts, rowptr, cntf);
  k_scatter<<<(N_EDGES + 255) / 256, 256, 0, stream>>>(src, dstv, rowptr, fill, epair);
  k_coord  <<<(N_EDGES + 255) / 256, 256, 0, stream>>>(pos, src, dstv, cdiff, ccross, cvert);

  // weight prep: concat + swizzle into WMMA fragment layout (bf16), once
  for (int l = 0; l < 4; ++l) {
    k_pack_qkvs_f32<<<(HID * QKVS + 255) / 256, 256, 0, stream>>>(
        GAT(l,0), GAT(l,2), GAT(l,4), GAT(l,7),
        GAT(l,1), GAT(l,3), GAT(l,5), GAT(l,8), wtmp, bqkvs[l]);
    k_swizzle_w<<<((HID/32)*(QKVS/16)*32 + 255) / 256, 256, 0, stream>>>(wtmp, wqkvs[l], HID, QKVS);
    k_swizzle_w<<<((HID/32)*(HID/16)*32 + 255) / 256, 256, 0, stream>>>(GAT(l,6),  wwe[l], HID, HID);
    k_swizzle_w<<<((HID/32)*(HID/16)*32 + 255) / 256, 256, 0, stream>>>(GAT(l,9),  ww1[l], HID, HID);
    k_swizzle_w<<<((HID/32)*(HID/16)*32 + 255) / 256, 256, 0, stream>>>(GAT(l,11), ww2[l], HID, HID);
  }
  for (int m = 0; m < 2; ++m)
    k_swizzle_w<<<((2*HID/32)*(COFF/16)*32 + 255) / 256, 256, 0, stream>>>(
        BAS(m,0), wb1[m], 2 * HID, COFF);

  hipMemcpyAsync(xbuf, node_attr, (size_t)N_NODES * HID * 4, hipMemcpyDeviceToDevice, stream);

  const dim3 gw(32, 4);
  // wave-count grids: (row_tiles * tile_groups + 3) / 4 blocks of 4 waves
  const int g_qkvs  = ((N_NODES / 16) * (QKVS / 64) + 3) / 4;
  const int g_eproj = ((N_EDGES / 16) * (HID  / 64) + 3) / 4;
  const int g_ffn   = ((N_NODES / 16) * (HID  / 64) + 3) / 4;
  for (int m = 0; m < 2; ++m) {
    for (int cix = 0; cix < 2; ++cix) {
      const int l = m * 2 + cix;
      // fused q|k|v|skip node GEMM, [N,128] @ [128,512]
      k_gemm_t<HID, QKVS><<<g_qkvs, gw, 0, stream>>>(xbuf, wqkvs[l], bqkvs[l], qkvs, N_NODES, 0);
      // edge projection, [E,128] @ [128,128] (no bias)
      k_gemm_t<HID, HID><<<g_eproj, gw, 0, stream>>>(edge_attr, wwe[l], nullptr, eproj, N_EDGES, 0);
      // attention: one wave/node, online softmax over CSR edges
      k_attn   <<<(N_NODES + 7) / 8, dim3(32, 8), 0, stream>>>(qkvs, eproj, rowptr, epair, hbuf);
      k_addnorm<<<(N_NODES + 7) / 8, dim3(32, 8), 0, stream>>>(xbuf, hbuf, GAT(l,13), GAT(l,14), 0);
      // FFN
      k_gemm_t<HID, HID><<<g_ffn, gw, 0, stream>>>(xbuf, ww1[l], GAT(l,10), tbuf, N_NODES, 1);
      k_gemm_t<HID, HID><<<g_ffn, gw, 0, stream>>>(tbuf, ww2[l], GAT(l,12), hbuf, N_NODES, 0);
      k_addnorm<<<(N_NODES + 7) / 8, dim3(32, 8), 0, stream>>>(
          xbuf, hbuf, GAT(l,15), GAT(l,16), (cix == 0) ? 1 : 0);   // inter-conv silu
    }
    // basis MLP: fused gather-GEMM -> eproj buffer holds ehid [E,64]
    k_basis_gemm<<<N_EDGES / 64, gw, 0, stream>>>(
        xbuf, edge_attr, src, dstv, wb1[m], BAS(m,1), eproj);
    k_basis_out<<<(N_EDGES + 255) / 256, 256, 0, stream>>>(
        eproj, BAS(m,2), BAS(m,3), cdiff, ccross, cvert, dstv, grad);
  }

  // outputs: x then gradient/cnt
  hipMemcpyAsync(d_out, xbuf, (size_t)N_NODES * HID * 4, hipMemcpyDeviceToDevice, stream);
  k_gradout<<<(N_NODES * 3 + 255) / 256, 256, 0, stream>>>(
      grad, cntf, (float*)d_out + (size_t)N_NODES * HID);
#undef GAT
#undef BAS
}
